// HierarchicalReasoningProjector_13056700580094
// MI455X (gfx1250) — matrix operand order repaired and checked by reading, not compiled, not run on hardware
//
#include <hip/hip_runtime.h>
#include <hip/hip_bf16.h>
#include <math.h>

#define Dm 2048
#define Tm 1024
#define Bm 16
#define NHm 4
#define HDm 512
#define CHUNK 128  // K-chunk staged through LDS in the WMMA GEMM

#define AS1 __attribute__((address_space(1)))
#define AS3 __attribute__((address_space(3)))

typedef __attribute__((ext_vector_type(16))) _Float16 v16h;
typedef __attribute__((ext_vector_type(8)))  float    v8f;
typedef __attribute__((__vector_size__(4 * sizeof(int)))) int i32x4;  // async-LDS payload type

__device__ __forceinline__ uint32_t pack2h(float a, float b) {
  union { _Float16 h[2]; uint32_t u; } p;
  p.h[0] = (_Float16)a; p.h[1] = (_Float16)b;
  return p.u;
}

__device__ __forceinline__ float wave_sum(float v) {
  #pragma unroll
  for (int off = 16; off > 0; off >>= 1) v += __shfl_down(v, off, 32);
  return v;
}

__device__ __forceinline__ float gelu_exact(float v) {
  return 0.5f * v * (1.0f + erff(v * 0.70710678118654752f));
}

__device__ __forceinline__ float dot4(float4 a, float4 b) {
  return a.x * b.x + a.y * b.y + a.z * b.z + a.w * b.w;
}

// ---------------------------------------------------------------------------
// Generic vector helpers
// ---------------------------------------------------------------------------
__global__ void zero_kernel(float* __restrict__ p, int n) {
  int i = blockIdx.x * 256 + threadIdx.x;
  if (i < n) p[i] = 0.0f;
}

__global__ void vec_add_kernel(float* __restrict__ dst, const float* __restrict__ a,
                               const float* __restrict__ b, int n) {
  int i = blockIdx.x * 256 + threadIdx.x;
  if (i < n) dst[i] = a[i] + b[i];
}

// LayerNorm over a single 2048-vector (one workgroup).
__global__ void ln_kernel(float* __restrict__ dst, const float* __restrict__ src,
                          const float* __restrict__ g, const float* __restrict__ b) {
  __shared__ float s1[256], s2[256];
  int t = threadIdx.x;
  float a1 = 0.f, a2 = 0.f;
  for (int i = t; i < Dm; i += 256) { float x = src[i]; a1 += x; a2 += x * x; }
  s1[t] = a1; s2[t] = a2; __syncthreads();
  for (int off = 128; off > 0; off >>= 1) {
    if (t < off) { s1[t] += s1[t + off]; s2[t] += s2[t + off]; }
    __syncthreads();
  }
  float mean = s1[0] * (1.0f / Dm);
  float var  = s2[0] * (1.0f / Dm) - mean * mean;
  float rs   = rsqrtf(var + 1e-5f);
  for (int i = t; i < Dm; i += 256) dst[i] = (src[i] - mean) * rs * g[i] + b[i];
}

// GEMV: dst[n] = [resid[n] +] [gelu](dot(W[n,:], v) [+ bias[n]])
// One wave per output row; float4 coalesced streaming (L2-resident weights).
__global__ void gemv_kernel(float* __restrict__ dst, const float* __restrict__ W,
                            const float* __restrict__ v, const float* __restrict__ bias,
                            const float* __restrict__ resid, int N, int K, int apply_gelu) {
  int wid = threadIdx.x >> 5, lane = threadIdx.x & 31;
  int n = blockIdx.x * 8 + wid;
  if (n >= N) return;
  const float4* W4 = (const float4*)(W + (size_t)n * K);
  const float4* v4 = (const float4*)v;
  int nq = K >> 2;
  float acc = 0.f;
  for (int i = lane; i < nq; i += 32) acc += dot4(W4[i], v4[i]);
  acc = wave_sum(acc);
  if (lane == 0) {
    if (bias)  acc += bias[n];
    if (apply_gelu) acc = gelu_exact(acc);
    if (resid) acc += resid[n];
    dst[n] = acc;
  }
}

// ---------------------------------------------------------------------------
// Phase 1: collapsed cross-attention
// ---------------------------------------------------------------------------
// qk[h,:] = (wk_h^T q_h) / sqrt(HD)   (bias term is softmax-invariant, dropped)
__global__ void qk_kernel(float* __restrict__ qk, const float* __restrict__ in_w,
                          const float* __restrict__ qv) {
  int h = blockIdx.y;
  int j = blockIdx.x * 256 + threadIdx.x;
  const float* Wk = in_w + (size_t)(Dm + h * HDm) * Dm;
  const float* qh = qv + h * HDm;
  float acc = 0.f;
  for (int i = 0; i < HDm; ++i) acc += Wk[(size_t)i * Dm + j] * qh[i];
  qk[h * Dm + j] = acc * 0.044194173824159216f;  // 1/sqrt(512)
}

// scores[b,h,t] = qk[h,:] . x[b,t,:]   -- one streaming pass over x (128 MB)
__global__ void scores_kernel(float* __restrict__ scores, const float* __restrict__ x,
                              const float* __restrict__ qk) {
  __shared__ float qs[NHm * Dm];
  for (int i = threadIdx.x; i < NHm * Dm; i += 256) qs[i] = qk[i];
  __syncthreads();
  int wid = threadIdx.x >> 5, lane = threadIdx.x & 31;
  int row = blockIdx.x * 8 + wid;  // 0..16383 == b*1024+t
  const float4* xr = (const float4*)(x + (size_t)row * Dm);
  const float4* q0 = (const float4*)(qs);
  const float4* q1 = (const float4*)(qs + Dm);
  const float4* q2 = (const float4*)(qs + 2 * Dm);
  const float4* q3 = (const float4*)(qs + 3 * Dm);
  float a0 = 0.f, a1 = 0.f, a2 = 0.f, a3 = 0.f;
  for (int i = lane; i < Dm / 4; i += 32) {
    float4 xv = xr[i];
    __builtin_prefetch(xr + i + 128, 0, 1);  // global_prefetch_b8
    a0 += dot4(xv, q0[i]); a1 += dot4(xv, q1[i]);
    a2 += dot4(xv, q2[i]); a3 += dot4(xv, q3[i]);
  }
  a0 = wave_sum(a0); a1 = wave_sum(a1); a2 = wave_sum(a2); a3 = wave_sum(a3);
  if (lane == 0) {
    int b = row >> 10, t = row & 1023;
    float* s = scores + (size_t)b * NHm * Tm + t;
    s[0] = a0; s[Tm] = a1; s[2 * Tm] = a2; s[3 * Tm] = a3;
  }
}

// In-place softmax over 1024 logits per (b,h); one workgroup each.
__global__ void softmax_kernel(float* __restrict__ s) {
  float* p = s + (size_t)blockIdx.x * Tm;
  __shared__ float red[256];
  int t = threadIdx.x;
  float m = -1e30f;
  for (int i = t; i < Tm; i += 256) m = fmaxf(m, p[i]);
  red[t] = m; __syncthreads();
  for (int off = 128; off > 0; off >>= 1) {
    if (t < off) red[t] = fmaxf(red[t], red[t + off]);
    __syncthreads();
  }
  m = red[0]; __syncthreads();
  float sum = 0.f;
  for (int i = t; i < Tm; i += 256) { float e = expf(p[i] - m); p[i] = e; sum += e; }
  red[t] = sum; __syncthreads();
  for (int off = 128; off > 0; off >>= 1) {
    if (t < off) red[t] += red[t + off];
    __syncthreads();
  }
  float inv = 1.0f / red[0];
  for (int i = t; i < Tm; i += 256) p[i] *= inv;
}

// xbar[b,h,:] = sum_t att[b,h,t] * x[b,t,:]  -- second streaming pass over x
__global__ void xbar_kernel(float* __restrict__ xbar, const float* __restrict__ x,
                            const float* __restrict__ att) {
  int b = blockIdx.y;
  int d = blockIdx.x * 256 + threadIdx.x;
  __shared__ float as[NHm * 256];
  float acc0 = 0.f, acc1 = 0.f, acc2 = 0.f, acc3 = 0.f;
  for (int t0 = 0; t0 < Tm; t0 += 256) {
    __syncthreads();
    #pragma unroll
    for (int h = 0; h < NHm; ++h)
      as[h * 256 + threadIdx.x] = att[(size_t)b * NHm * Tm + h * Tm + t0 + threadIdx.x];
    __syncthreads();
    for (int tt = 0; tt < 256; ++tt) {
      float xv = x[((size_t)b * Tm + t0 + tt) * Dm + d];
      acc0 += as[tt] * xv;       acc1 += as[256 + tt] * xv;
      acc2 += as[512 + tt] * xv; acc3 += as[768 + tt] * xv;
    }
  }
  float* o = xbar + (size_t)b * NHm * Dm + d;
  o[0] = acc0; o[Dm] = acc1; o[2 * Dm] = acc2; o[3 * Dm] = acc3;
}

// ---------------------------------------------------------------------------
// WMMA GEMM, M = 16:  C[16 x N] = A[16 x K] * W[N x K]^T  (+bias, gelu, resid)
// One wave per 16x16 output tile; V_WMMA_F32_16X16X32_F16.
// Both A and the per-wave W tiles are staged into LDS with
// GLOBAL_LOAD_ASYNC_TO_LDS_B128 (ASYNCcnt) when the toolchain exposes the
// builtin, with a coalesced float4 fallback otherwise. Fragments are then
// built from LDS only (no strided global traffic in the inner loop).
// ---------------------------------------------------------------------------
__global__ void wmma_gemm16_kernel(float* __restrict__ C, int ldc, int col0,
                                   const float* __restrict__ A, int lda,
                                   const float* __restrict__ W,
                                   const float* __restrict__ bias,
                                   const float* __restrict__ resid,
                                   int K, int apply_gelu) {
  __shared__ float    aF[16 * CHUNK];      //  8 KB raw f32 A chunk (shared by block)
  __shared__ float    wF[4][16 * CHUNK];   // 32 KB raw f32 W tiles (one per wave)
  __shared__ uint32_t aH[16 * (CHUNK / 2)];//  4 KB A packed to f16 pairs

  int tid = threadIdx.x;
  int wid = tid >> 5, lane = tid & 31;
  int n0 = (blockIdx.x * 4 + wid) * 16;
  int m = lane & 15, c = lane >> 4;
  v8f acc = {0.f, 0.f, 0.f, 0.f, 0.f, 0.f, 0.f, 0.f};

  for (int kc = 0; kc < K; kc += CHUNK) {
    __syncthreads();  // previous chunk fully consumed before overwrite
#if __has_builtin(__builtin_amdgcn_global_load_async_to_lds_b128)
    {
      // A tile: 16 rows x CHUNK f32 = 512 x b128 transfers, all 128 threads.
      AS3 i32x4* ldsA = (AS3 i32x4*)(uint32_t)(uintptr_t)&aF[0];
      for (int i = tid; i < 16 * CHUNK / 4; i += 128) {
        int row = i / (CHUNK / 4), seg = i % (CHUNK / 4);
        __builtin_amdgcn_global_load_async_to_lds_b128(
            (AS1 i32x4*)(uintptr_t)(A + (size_t)row * lda + kc + seg * 4),
            ldsA + i, 0, 0);
      }
      // W tile for this wave: 512 x b128 transfers across the wave's 32 lanes.
      AS3 i32x4* ldsW = (AS3 i32x4*)(uint32_t)(uintptr_t)&wF[wid][0];
      for (int i = lane; i < 16 * CHUNK / 4; i += 32) {
        int row = i / (CHUNK / 4), seg = i % (CHUNK / 4);
        __builtin_amdgcn_global_load_async_to_lds_b128(
            (AS1 i32x4*)(uintptr_t)(W + (size_t)(n0 + row) * K + kc + seg * 4),
            ldsW + i, 0, 0);
      }
#if __has_builtin(__builtin_amdgcn_s_wait_asynccnt)
      __builtin_amdgcn_s_wait_asynccnt(0);
#else
      asm volatile("s_wait_asynccnt 0x0" ::: "memory");
#endif
    }
#else
    for (int i = tid; i < 16 * CHUNK / 4; i += 128) {
      int row = i / (CHUNK / 4), seg = i % (CHUNK / 4);
      ((float4*)aF)[i] = *(const float4*)(A + (size_t)row * lda + kc + seg * 4);
    }
    for (int i = lane; i < 16 * CHUNK / 4; i += 32) {
      int row = i / (CHUNK / 4), seg = i % (CHUNK / 4);
      ((float4*)&wF[wid][0])[i] = *(const float4*)(W + (size_t)(n0 + row) * K + kc + seg * 4);
    }
#endif
    __syncthreads();
    // Pack A f32 -> f16 pairs (fragment-friendly contiguous uints).
    for (int p = tid; p < 16 * (CHUNK / 2); p += 128)
      aH[p] = pack2h(aF[2 * p], aF[2 * p + 1]);
    __syncthreads();

    for (int ks = 0; ks < CHUNK; ks += 32) {
      union { v16h h; uint32_t u[8]; } af, bf;
      // A fragment (16x32 f16): lane holds row m; K pairs per ISA layout.
      int base = m * (CHUNK / 2) + (ks >> 1) + 4 * c;
      #pragma unroll
      for (int j = 0; j < 4; ++j) {
        af.u[j]     = aH[base + j];
        af.u[4 + j] = aH[base + 8 + j];
      }
      // B fragment (32x16 f16): lane holds column n = lane%16; K pairs per ISA layout (same as A).
      const float* wcol = &wF[wid][(lane & 15) * CHUNK + ks + 8 * c];
      #pragma unroll
      for (int j = 0; j < 4; ++j) {
        bf.u[j]     = pack2h(wcol[2 * j], wcol[2 * j + 1]);
        bf.u[4 + j] = pack2h(wcol[16 + 2 * j], wcol[17 + 2 * j]);
      }
      acc = __builtin_amdgcn_wmma_f32_16x16x32_f16(
          false, af.h, false, bf.h, (short)0, acc, false, false);
    }
  }
  int n = n0 + (lane & 15);
  #pragma unroll
  for (int r = 0; r < 8; ++r) {
    int mo = r + 8 * c;  // C/D layout: VGPR r -> row r (+8 for hi lanes)
    float vv = acc[r];
    if (bias)  vv += bias[n];
    if (apply_gelu) vv = gelu_exact(vv);
    if (resid) vv += resid[(size_t)mo * ldc + col0 + n];
    C[(size_t)mo * ldc + col0 + n] = vv;
  }
}

// ---------------------------------------------------------------------------
extern "C" void kernel_launch(void* const* d_in, const int* in_sizes, int n_in,
                              void* d_out, int out_size, void* d_ws, size_t ws_size,
                              hipStream_t stream) {
  (void)in_sizes; (void)n_in; (void)out_size; (void)ws_size;
  const float* x     = (const float*)d_in[0];
  const float* gq    = (const float*)d_in[1];
  const float* in_w  = (const float*)d_in[2];
  const float* in_b  = (const float*)d_in[3];
  const float* out_w = (const float*)d_in[4];
  const float* out_b = (const float*)d_in[5];
  const float* m1w   = (const float*)d_in[6];
  const float* m1b   = (const float*)d_in[7];
  const float* m2w   = (const float*)d_in[8];
  const float* m2b   = (const float*)d_in[9];
  // H / L block params (seq-len-1 attention: only the V slice of in_proj matters)
  const float* hP[12]; const float* lP[12];
  for (int i = 0; i < 12; ++i) { hP[i] = (const float*)d_in[10 + i]; lP[i] = (const float*)d_in[22 + i]; }
  const float* norm_g = (const float*)d_in[34];
  const float* norm_b = (const float*)d_in[35];

  float* ws  = (float*)d_ws;
  float* qv  = ws;                 // 2048
  float* qks = qv  + Dm;           // 4*2048
  float* sc  = qks + NHm * Dm;     // 16*4*1024
  float* xb  = sc  + Bm * NHm * Tm;         // 16*4*2048
  float* ov  = xb  + Bm * NHm * Dm;         // 16*2048
  float* a0  = ov  + Bm * Dm;               // 16*2048
  float* a1  = a0  + Bm * Dm;               // 16*2048
  float* av  = a1  + Bm * Dm;               // 16*2048  (scan input rows)
  float* zH  = av  + Bm * Dm;               // 2048
  float* zL  = zH  + Dm;                    // 2048 (contiguous with zH)
  float* inj = zL  + Dm;                    // 2048
  float* xv  = inj + Dm;                    // 2048
  float* n1v = xv  + Dm;                    // 2048
  float* t1  = n1v + Dm;                    // 2048

  // ---- Phase 1: collapsed cross-attention + MLP (bandwidth: 2 passes of x)
  gemv_kernel<<<Dm / 8, 256, 0, stream>>>(qv, in_w, gq, in_b, nullptr, Dm, Dm, 0);
  qk_kernel<<<dim3(Dm / 256, NHm), 256, 0, stream>>>(qks, in_w, qv);
  scores_kernel<<<(Bm * Tm) / 8, 256, 0, stream>>>(sc, x, qks);
  softmax_kernel<<<Bm * NHm, 256, 0, stream>>>(sc);
  xbar_kernel<<<dim3(Dm / 256, Bm), 256, 0, stream>>>(xb, x, sc);
  for (int h = 0; h < NHm; ++h)  // o[:, h*512:(h+1)*512] = xbar_h @ wv_h^T + bv_h
    wmma_gemm16_kernel<<<HDm / 64, 128, 0, stream>>>(
        ov, Dm, h * HDm, xb + h * Dm, NHm * Dm,
        in_w + (size_t)(2 * Dm + h * HDm) * Dm, in_b + 2 * Dm + h * HDm,
        nullptr, Dm, 0);
  wmma_gemm16_kernel<<<Dm / 64, 128, 0, stream>>>(a0, Dm, 0, ov, Dm, out_w, out_b, nullptr, Dm, 0);
  wmma_gemm16_kernel<<<Dm / 64, 128, 0, stream>>>(a1, Dm, 0, a0, Dm, m1w, m1b, nullptr, Dm, 1);
  wmma_gemm16_kernel<<<Dm / 64, 128, 0, stream>>>(av, Dm, 0, a1, Dm, m2w, m2b, nullptr, Dm, 0);

  // ---- Phase 2: sequential scan (64 blocks of M=1 GEMVs, L2-resident weights)
  zero_kernel<<<16, 256, 0, stream>>>(zH, 2 * Dm);

  auto run_block = [&](float* z, const float* injv, const float* const* P) {
    const float* wv  = P[0] + (size_t)2 * Dm * Dm;  // V slice of in_proj
    const float* bv  = P[1] + 2 * Dm;
    const float* ow  = P[2];  const float* ob  = P[3];
    const float* n1g = P[4];  const float* n1b = P[5];
    const float* n2g = P[6];  const float* n2b = P[7];
    const float* f1w = P[8];  const float* f1b = P[9];
    const float* f2w = P[10]; const float* f2b = P[11];
    vec_add_kernel<<<Dm / 256, 256, 0, stream>>>(xv, z, injv, Dm);
    ln_kernel<<<1, 256, 0, stream>>>(n1v, xv, n1g, n1b);
    gemv_kernel<<<Dm / 8, 256, 0, stream>>>(t1, wv, n1v, bv, nullptr, Dm, Dm, 0);
    gemv_kernel<<<Dm / 8, 256, 0, stream>>>(xv, ow, t1, ob, xv, Dm, Dm, 0);
    ln_kernel<<<1, 256, 0, stream>>>(n1v, xv, n2g, n2b);
    gemv_kernel<<<Dm / 8, 256, 0, stream>>>(t1, f1w, n1v, f1b, nullptr, Dm, Dm, 1);
    gemv_kernel<<<Dm / 8, 256, 0, stream>>>(z, f2w, t1, f2b, xv, Dm, Dm, 0);
  };

  for (int b = 0; b < Bm; ++b) {
    vec_add_kernel<<<Dm / 256, 256, 0, stream>>>(inj, zH, av + (size_t)b * Dm, Dm);
    run_block(zL, inj, lP);  // refinement 1 (zH frozen: loop-shadow bug)
    run_block(zL, inj, lP);  // refinement 2
    run_block(zL, inj, lP);  // grad-enabled L update
    run_block(zH, zL, hP);   // grad-enabled H update (x = zH + zL)
  }
  ln_kernel<<<1, 256, 0, stream>>>((float*)d_out, zH, norm_g, norm_b);
}
